// LinearRegion_30683246362895
// MI455X (gfx1250) — matrix operand order repaired and checked by reading, not compiled, not run on hardware
//
#include <hip/hip_runtime.h>
#include <hip/hip_bf16.h>

typedef __attribute__((ext_vector_type(16))) __bf16 v16bf;
typedef __attribute__((ext_vector_type(8)))  __bf16 v8bf;
typedef __attribute__((ext_vector_type(8)))  float  v8f;

#define BATCH   1024
#define NSITES  64
#define PDIM    4
#define DDIM    64
#define SMID    62                    // middle sites
#define SITE_ELTS (PDIM*DDIM*DDIM)    // 16384 elements per site
#define BT      16                    // batch tile per workgroup
#define VPADB   72                    // LDS row pad (bf16 elts): 144B rows, 16B-aligned runs
#define WS_X_BYTE_OFF ((size_t)SMID * SITE_ELTS * 2)   // bf16 cores region size (16B aligned)

// ---------------------------------------------------------------------------
// Pre-pass 1: convert mid_cores f32 -> bf16 and permute into the per-lane
// B-fragment layout for v_wmma_f32_16x16x32_bf16:
//   out[s][kc][t][lane][e]  with  K = kc*32 + e + 16*(lane/16),  r = t*16 + lane%16
// kc = i*2 + kc2  (i = physical leg, kc2 = K-half within C_i).
// ---------------------------------------------------------------------------
__global__ __launch_bounds__(256)
void convert_cores_kernel(const float* __restrict__ mid, __bf16* __restrict__ wsb) {
    int idx = blockIdx.x * 256 + threadIdx.x;
    if (idx >= SMID * SITE_ELTS) return;
    int s   = idx >> 14;           // /16384
    int rem = idx & 16383;
    int l   = rem >> 8;            // source layout [s][l][i][r]
    int i   = (rem >> 6) & 3;
    int r   = rem & 63;
    float v = mid[idx];

    int K  = i * 64 + l;           // GEMM K index (i-major)
    int kc = K >> 5;               // chunk of 32 (== i*2 + l/32)
    int Kl = K & 31;
    int g  = Kl >> 4;              // lane half
    int e  = Kl & 15;              // element within fragment
    int t  = r >> 4;               // output N-tile (== wave id)
    int n  = r & 15;
    int out = s * SITE_ELTS + kc * 2048 + t * 512 + (g * 16 + n) * 16 + e;
    wsb[out] = (__bf16)v;
}

// ---------------------------------------------------------------------------
// Pre-pass 2: transpose x[b][site][i] -> xT[site][i][b]  (batch fastest) so
// the combine's (j, j+1) scalar pairs are register-adjacent => v_pk_fma_f32
// with no v_mov shuffling.
// ---------------------------------------------------------------------------
__global__ __launch_bounds__(256)
void transpose_x_kernel(const float* __restrict__ x, float* __restrict__ xT) {
    int idx = blockIdx.x * 256 + threadIdx.x;
    if (idx >= BATCH * NSITES * PDIM) return;
    int b    = idx >> 8;            // /256
    int site = (idx >> 2) & 63;
    int i    = idx & 3;
    xT[(site * PDIM + i) * BATCH + b] = x[idx];
}

// ---------------------------------------------------------------------------
// Main chain kernel: 64 WGs x 128 threads (4 wave32 waves).
// WG owns batches [b0, b0+16); wave t owns output columns r in [16t, 16t+16).
// Per site:  W_i = v @ C_i   (4 independent f32 accumulators, 2 WMMAs each,
//            A-fragments = raw bf16 v straight from LDS, zero VALU)
//            v_new[b,r] = sum_i xT[s+1][i][b] * W_i[b,r]  (packed f32 FMAs)
// ---------------------------------------------------------------------------
__global__ __launch_bounds__(128)
void mps_chain_kernel(const float* __restrict__ x,
                      const float* __restrict__ first,
                      const __bf16* __restrict__ wsb,
                      const float* __restrict__ xT,
                      const float* __restrict__ last,
                      float* __restrict__ out) {
    __shared__ __bf16 vbuf[2][BT * VPADB];

    const int tid  = threadIdx.x;
    const int wave = tid >> 5;     // 0..3  -> output N tile
    const int lane = tid & 31;
    const int g    = lane >> 4;    // lane half
    const int n    = lane & 15;
    const int b0   = blockIdx.x * BT;

    // ---- init v[b, r] = sum_i first[i, r] * x[b, 0, i]  (stored bf16) ----
    {
        int m  = tid >> 3;                 // 0..15
        int r0 = (tid & 7) * 8;            // 8 columns each
        const float* xr = x + (size_t)(b0 + m) * (NSITES * PDIM); // site 0
        float x0 = xr[0], x1 = xr[1], x2 = xr[2], x3 = xr[3];
#pragma unroll
        for (int rr = 0; rr < 8; ++rr) {
            int r = r0 + rr;
            vbuf[0][m * VPADB + r] = (__bf16)(
                x0 * first[r] + x1 * first[64 + r] + x2 * first[128 + r] + x3 * first[192 + r]);
        }
    }
    __syncthreads();

    int cur = 0;
    for (int s = 0; s < SMID; ++s) {
        const __bf16* vcur = &vbuf[cur][0];
        __bf16*       vnxt = &vbuf[cur ^ 1][0];

        // xT[s+1][i][b0+8g+j], j=0..7: per i, two float4s -> (j,j+1) pairs adjacent
        float xv[4][8];
        {
            const float* xs = xT + (size_t)(s + 1) * (PDIM * BATCH) + b0 + 8 * g;
#pragma unroll
            for (int i = 0; i < 4; ++i) {
                *(float4*)&xv[i][0] = *(const float4*)(xs + i * BATCH);
                *(float4*)&xv[i][4] = *(const float4*)(xs + i * BATCH + 4);
            }
        }

        // A fragments: raw bf16 v rows from LDS, two 16B runs per chunk (no VALU)
        v16bf afrag[2];
#pragma unroll
        for (int kc2 = 0; kc2 < 2; ++kc2) {
            const __bf16* row = vcur + n * VPADB + kc2 * 32;
            v8bf lo = *(const v8bf*)(row + 16 * g);       // e=0..7  -> K_local = e + 16g
            v8bf hi = *(const v8bf*)(row + 16 * g + 8);   // e=8..15 -> K_local = e + 16g
            afrag[kc2] = __builtin_shufflevector(lo, hi,
                0, 1, 2, 3, 4, 5, 6, 7, 8, 9, 10, 11, 12, 13, 14, 15);
        }

        const __bf16* siteB = wsb + (size_t)s * SITE_ELTS + wave * 512 + lane * 16;

        // four independent accumulators: W_i = v @ C_i
        v8f acc0 = {}, acc1 = {}, acc2 = {}, acc3 = {};
#pragma unroll
        for (int kc2 = 0; kc2 < 2; ++kc2) {
            v16bf b0f = *(const v16bf*)(siteB + (0 * 2 + kc2) * 2048);
            acc0 = __builtin_amdgcn_wmma_f32_16x16x32_bf16(false, afrag[kc2], false, b0f, (short)0, acc0, false, false);
            v16bf b1f = *(const v16bf*)(siteB + (1 * 2 + kc2) * 2048);
            acc1 = __builtin_amdgcn_wmma_f32_16x16x32_bf16(false, afrag[kc2], false, b1f, (short)0, acc1, false, false);
            v16bf b2f = *(const v16bf*)(siteB + (2 * 2 + kc2) * 2048);
            acc2 = __builtin_amdgcn_wmma_f32_16x16x32_bf16(false, afrag[kc2], false, b2f, (short)0, acc2, false, false);
            v16bf b3f = *(const v16bf*)(siteB + (3 * 2 + kc2) * 2048);
            acc3 = __builtin_amdgcn_wmma_f32_16x16x32_bf16(false, afrag[kc2], false, b3f, (short)0, acc3, false, false);
        }

        // prefetch next site's 32KB B block
        if (s + 1 < SMID) {
            const char* nb = (const char*)(wsb + (size_t)(s + 1) * SITE_ELTS);
            __builtin_prefetch(nb + tid * 256, 0, 1);
            __builtin_prefetch(nb + tid * 256 + 128, 0, 1);
        }

        // combine: v_new[row j+8g, col 16*wave+n] = sum_i x_i * W_i  (f32), store bf16
#pragma unroll
        for (int j = 0; j < 8; ++j) {
            float cj = xv[0][j] * acc0[j] + xv[1][j] * acc1[j]
                     + xv[2][j] * acc2[j] + xv[3][j] * acc3[j];
            vnxt[(j + 8 * g) * VPADB + wave * 16 + n] = (__bf16)cj;
        }
        __syncthreads();
        cur ^= 1;
    }

    // ---- epilogue: out[b] = sum_l v[b,l] * (sum_i last[l,i] * x[b,N-1,i]) ----
    if (tid < BT) {
        int m = tid, b = b0 + m;
        const float* xl = x + (size_t)b * (NSITES * PDIM) + (NSITES - 1) * PDIM;
        float x0 = xl[0], x1 = xl[1], x2 = xl[2], x3 = xl[3];
        const __bf16* vrow = &vbuf[cur][m * VPADB];
        float acc = 0.f;
#pragma unroll 4
        for (int l = 0; l < DDIM; ++l) {
            float lm = last[l * 4 + 0] * x0 + last[l * 4 + 1] * x1 +
                       last[l * 4 + 2] * x2 + last[l * 4 + 3] * x3;
            acc += (float)vrow[l] * lm;
        }
        out[b] = acc;
    }
}

// ---------------------------------------------------------------------------
extern "C" void kernel_launch(void* const* d_in, const int* in_sizes, int n_in,
                              void* d_out, int out_size, void* d_ws, size_t ws_size,
                              hipStream_t stream) {
    const float* x     = (const float*)d_in[0];  // [B, N, P]
    const float* first = (const float*)d_in[1];  // [P, D]
    const float* mid   = (const float*)d_in[2];  // [N-2, D, P, D]
    const float* last  = (const float*)d_in[3];  // [D, P]
    float* out = (float*)d_out;                  // [B, 1]

    __bf16* wsb = (__bf16*)d_ws;                               // ~1.94MB bf16 cores
    float*  xT  = (float*)((char*)d_ws + WS_X_BYTE_OFF);       // 1MB transposed x

    const int totalC = SMID * SITE_ELTS;
    convert_cores_kernel<<<(totalC + 255) / 256, 256, 0, stream>>>(mid, wsb);
    const int totalX = BATCH * NSITES * PDIM;
    transpose_x_kernel<<<(totalX + 255) / 256, 256, 0, stream>>>(x, xT);
    mps_chain_kernel<<<BATCH / BT, 128, 0, stream>>>(x, first, wsb, xT, last, out);
}
